// SwinTransformerBlock_87582973100286
// MI455X (gfx1250) — compile-verified, matrix operand order checked by reading, not run on hardware
//
#include <hip/hip_runtime.h>
#include <hip/hip_bf16.h>
#include <math.h>

typedef __bf16 bf16;
typedef bf16  v16bf __attribute__((ext_vector_type(16)));
typedef float v8f   __attribute__((ext_vector_type(8)));
typedef int   v4i   __attribute__((ext_vector_type(4)));

#define TOKENS 200704        // 64 * 56 * 56
#define SCALE_ATTN 0.17677669529663689f   // 32^-0.5

#if defined(__gfx1250__) && __has_builtin(__builtin_amdgcn_global_load_async_to_lds_b128)
#define HAVE_ASYNC_LDS 1
typedef __attribute__((address_space(1))) v4i* gptr_v4i;   // global b128 granule
typedef __attribute__((address_space(3))) v4i* lptr_v4i;   // LDS b128 granule
#else
#define HAVE_ASYNC_LDS 0
#endif

__device__ inline void wait_asynccnt0() {
#if HAVE_ASYNC_LDS
#if __has_builtin(__builtin_amdgcn_s_wait_asynccnt)
  __builtin_amdgcn_s_wait_asynccnt(0);
#else
  asm volatile("s_wait_asynccnt 0x0" ::: "memory");
#endif
#endif
}

union FragU { v16bf v; unsigned u[8]; };

// Load a 16x32 bf16 WMMA operand fragment from LDS.
// A-operand: idx0 = tile row base, lane&15 selects row, halves split K.
// B-operand: identical packing with idx0 = tile col base (B stored transposed).
// Per ISA 7.12.2: element pair j (vgpr j) at k = (j>=4?16:0) + half*8 + (j&3)*2.
__device__ inline v16bf ldsFrag(const bf16* base, int stride, int idx0, int k0, int lane) {
  const int row = idx0 + (lane & 15);
  const int hlf = (lane >> 4) & 1;
  const bf16* p = base + row * stride + k0;
  FragU f;
#pragma unroll
  for (int j = 0; j < 8; ++j) {
    int k = ((j >> 2) << 4) + (hlf << 3) + ((j & 3) << 1);
    f.u[j] = *(const unsigned*)(p + k);   // 2 bf16, 4B aligned (k even, stride even)
  }
  return f.v;
}

__device__ inline int shift_region(int p) {   // Swin shift-mask region id along one axis
  return (p < 49) ? 0 : ((p < 53) ? 1 : 2);   // H-WS=49, H-SHIFT=53
}

// -------------------------------------------------------------------------
// K1/K5a: LayerNorm (fp32) -> bf16, optionally fused with cyclic shift +
// window partition (token reordering). One wave (32 lanes) per token.
// -------------------------------------------------------------------------
__global__ __launch_bounds__(256)
void ln_kernel(const float* __restrict__ x, const float* __restrict__ g,
               const float* __restrict__ beta, bf16* __restrict__ out,
               int windowize)
{
  const int tid  = threadIdx.x;
  const int lane = tid & 31;
  const int wave = tid >> 5;
  const int o = blockIdx.x * 8 + wave;   // output token slot (window order if windowize)

  size_t src;
  if (windowize) {
    int win = o / 49, n = o % 49;
    int b = win >> 6, wimg = win & 63;
    int wy = wimg >> 3, wx = wimg & 7;
    int iy = n / 7, ix = n % 7;
    int hs = wy * 7 + iy + 3; if (hs >= 56) hs -= 56;   // roll(-SHIFT): shifted[i]=x[(i+3)%56]
    int wc = wx * 7 + ix + 3; if (wc >= 56) wc -= 56;
    src = (size_t)b * 3136 + hs * 56 + wc;
  } else {
    src = (size_t)o;
  }

  const float* xr = x + src * 256 + lane * 8;
  float vals[8];
  float4 a0 = *(const float4*)(xr);
  float4 a1 = *(const float4*)(xr + 4);
  vals[0]=a0.x; vals[1]=a0.y; vals[2]=a0.z; vals[3]=a0.w;
  vals[4]=a1.x; vals[5]=a1.y; vals[6]=a1.z; vals[7]=a1.w;

  float s = 0.f, sq = 0.f;
#pragma unroll
  for (int i = 0; i < 8; ++i) { s += vals[i]; sq += vals[i] * vals[i]; }
#pragma unroll
  for (int m = 16; m >= 1; m >>= 1) {
    s  += __shfl_xor(s,  m, 32);
    sq += __shfl_xor(sq, m, 32);
  }
  float mean = s * (1.f / 256.f);
  float var  = sq * (1.f / 256.f) - mean * mean;
  float inv  = rsqrtf(var + 1e-5f);

  bf16* orow = out + (size_t)o * 256 + lane * 8;
#pragma unroll
  for (int i = 0; i < 8; ++i) {
    int c = lane * 8 + i;
    float yv = (vals[i] - mean) * inv * g[c] + beta[c];
    orow[i] = (bf16)yv;
  }
}

// -------------------------------------------------------------------------
// Generic bf16 WMMA GEMM: C[M,N] = A[M,K](bf16) * B[K,N](fp32 weights,
// converted to bf16 in LDS) with fused epilogues.
// 256 threads = 8 waves; block tile 128x64; wave tile 32x32 (2x2 WMMA).
// A tile staged via GLOBAL_LOAD_ASYNC_TO_LDS_B128 (ASYNCcnt) when available;
// B tile goes through registers (fp32->bf16 convert + transpose).
// EPI 0: +b_qkv, scatter to [win,3,head,n,hd] bf16
// EPI 1: +b_proj, window-reverse + unshift + residual, fp32 -> outf
// EPI 2: +b_mlp1, exact GELU, bf16 row-major
// EPI 3: +b_mlp2, outf[m*256+j] += v (residual accumulate)
// -------------------------------------------------------------------------
template<int EPI>
__global__ __launch_bounds__(256)
void gemm_bf16_kernel(const bf16* __restrict__ A, const float* __restrict__ Bw,
                      const float* __restrict__ bias, bf16* __restrict__ outb,
                      float* __restrict__ outf, const float* __restrict__ resid,
                      int M, int N, int K)
{
  constexpr int BM = 128, BN = 64, BK = 32;
  constexpr int AP = BK + 8;               // 80B row stride: 16B-aligned for async b128,
                                           // 20-bank step -> conflict-free frag gathers
  __shared__ __align__(16) bf16 As[BM][AP];
  __shared__ bf16 Bst[BN][BK + 2];         // B stored transposed: Bst[n][k]

  const int tid   = threadIdx.x;
  const int lane  = tid & 31;
  const int wave  = tid >> 5;
  const int waveM = wave & 3;              // 4 waves over M
  const int waveN = wave >> 2;             // 2 waves over N
  const int m0 = blockIdx.x * BM;
  const int n0 = blockIdx.y * BN;

  const v8f vzero = {0.f,0.f,0.f,0.f,0.f,0.f,0.f,0.f};
  v8f acc[2][2];
#pragma unroll
  for (int s = 0; s < 2; ++s)
#pragma unroll
    for (int t = 0; t < 2; ++t) acc[s][t] = vzero;

  const int kTiles = K / BK;
  for (int kt = 0; kt < kTiles; ++kt) {
    __syncthreads();
    // ---- stage A tile: 128x32 bf16, 16B granules ----
#pragma unroll
    for (int it = 0; it < 2; ++it) {
      int slot = it * 256 + tid;
      int row  = slot >> 2;
      int gk   = (slot & 3) << 3;
      const bf16* gsrc = A + (size_t)(m0 + row) * K + kt * BK + gk;
#if HAVE_ASYNC_LDS
      __builtin_amdgcn_global_load_async_to_lds_b128(
          (gptr_v4i)gsrc, (lptr_v4i)&As[row][gk], 0, 0);
#else
      const uint4 d = *(const uint4*)gsrc;
      unsigned* dst = (unsigned*)&As[row][gk];
      dst[0] = d.x; dst[1] = d.y; dst[2] = d.z; dst[3] = d.w;
#endif
    }
    if (kt + 1 < kTiles)   // gfx1250 global_prefetch for the next A k-tile
      __builtin_prefetch(A + (size_t)(m0 + (tid >> 1)) * K + (kt + 1) * BK, 0, 1);
    // ---- stage B tile: 32x64 fp32 -> bf16, transposed into Bst ----
#pragma unroll
    for (int it = 0; it < 2; ++it) {
      int slot = it * 256 + tid;
      int kk   = slot >> 4;
      int nn   = (slot & 15) << 2;
      const float4 wv = *(const float4*)(Bw + (size_t)(kt * BK + kk) * N + n0 + nn);
      Bst[nn + 0][kk] = (bf16)wv.x;
      Bst[nn + 1][kk] = (bf16)wv.y;
      Bst[nn + 2][kk] = (bf16)wv.z;
      Bst[nn + 3][kk] = (bf16)wv.w;
    }
    wait_asynccnt0();
    __syncthreads();

    v16bf af[2], bfr[2];
#pragma unroll
    for (int s = 0; s < 2; ++s)
      af[s] = ldsFrag(&As[0][0], AP, waveM * 32 + s * 16, 0, lane);
#pragma unroll
    for (int t = 0; t < 2; ++t)
      bfr[t] = ldsFrag(&Bst[0][0], BK + 2, waveN * 32 + t * 16, 0, lane);
#pragma unroll
    for (int s = 0; s < 2; ++s)
#pragma unroll
      for (int t = 0; t < 2; ++t)
        acc[s][t] = __builtin_amdgcn_wmma_f32_16x16x32_bf16(
            false, af[s], false, bfr[t], (short)0, acc[s][t], false, false);
  }

  // ---- epilogue: C/D layout VGPR e -> rows e / e+8 ----
  const int hlf = (lane >> 4) & 1;
  const int cL  = lane & 15;
#pragma unroll
  for (int s = 0; s < 2; ++s)
#pragma unroll
    for (int t = 0; t < 2; ++t)
#pragma unroll
      for (int e = 0; e < 8; ++e) {
        int m = m0 + waveM * 32 + s * 16 + e + hlf * 8;
        int j = n0 + waveN * 32 + t * 16 + cL;
        float v = acc[s][t][e] + bias[j];
        if (EPI == 0) {                      // qkv scatter
          int win = m / 49, n = m % 49;
          int which = j >> 8;
          int head  = (j >> 5) & 7;
          int hd    = j & 31;
          outb[(size_t)((((win * 3 + which) * 8 + head) * 49 + n) * 32 + hd)] = (bf16)v;
        } else if (EPI == 1) {               // proj + window reverse + residual
          int win = m / 49, n = m % 49;
          int b = win >> 6, wimg = win & 63;
          int wy = wimg >> 3, wx = wimg & 7;
          int iy = n / 7, ix = n % 7;
          int h = wy * 7 + iy + 3; if (h >= 56) h -= 56;
          int w = wx * 7 + ix + 3; if (w >= 56) w -= 56;
          size_t dst = ((size_t)b * 3136 + h * 56 + w) * 256 + j;
          outf[dst] = resid[dst] + v;
        } else if (EPI == 2) {               // exact GELU -> bf16 hidden
          float ge = 0.5f * v * (1.f + erff(v * 0.70710678118654752f));
          outb[(size_t)m * N + j] = (bf16)ge;
        } else {                             // mlp2 + residual accumulate
          size_t dst = (size_t)m * 256 + j;
          outf[dst] = outf[dst] + v;
        }
      }
}

// -------------------------------------------------------------------------
// K3: windowed attention. One workgroup (128 thr = 4 waves) per (window,head).
// S = Q*K^T (WMMA, HD=32 one K-step), fp32 softmax with inline rel-pos bias
// and shift mask, O = P*V (WMMA, K=64 two steps). 49 padded to 64 with zeros.
// -------------------------------------------------------------------------
__global__ __launch_bounds__(128)
void attn_kernel(const bf16* __restrict__ qkv, const float* __restrict__ bias_table,
                 bf16* __restrict__ out)
{
  __shared__ bf16 Qs[64][34];
  __shared__ bf16 Ks[64][34];
  __shared__ bf16 Vt[32][72];      // transposed: Vt[hd][token]
  __shared__ float Sf[64][65];
  __shared__ bf16 Ps[64][72];

  const int tid  = threadIdx.x;
  const int lane = tid & 31;
  const int wave = tid >> 5;
  const int win  = blockIdx.x >> 3;
  const int head = blockIdx.x & 7;
  const int wimg = win & 63;
  const int wy = wimg >> 3, wx = wimg & 7;

  const bf16* qp = qkv + (size_t)(((win * 3 + 0) * 8 + head)) * (49 * 32);
  const bf16* kp = qkv + (size_t)(((win * 3 + 1) * 8 + head)) * (49 * 32);
  const bf16* vp = qkv + (size_t)(((win * 3 + 2) * 8 + head)) * (49 * 32);

  for (int i = tid; i < 64 * 32; i += 128) {
    int n = i >> 5, hd = i & 31;
    bf16 zq = (n < 49) ? qp[n * 32 + hd] : (bf16)0.f;
    bf16 zk = (n < 49) ? kp[n * 32 + hd] : (bf16)0.f;
    bf16 zv = (n < 49) ? vp[n * 32 + hd] : (bf16)0.f;
    Qs[n][hd] = zq;
    Ks[n][hd] = zk;
    Vt[hd][n] = zv;
  }
  __syncthreads();

  const v8f vzero = {0.f,0.f,0.f,0.f,0.f,0.f,0.f,0.f};
  const int hlf = (lane >> 4) & 1;
  const int cL  = lane & 15;

  // ---- S = Q * K^T : wave w owns rows [16w,16w+16), 4 col tiles ----
  v16bf qa = ldsFrag(&Qs[0][0], 34, wave * 16, 0, lane);
  v8f sacc[4];
#pragma unroll
  for (int t = 0; t < 4; ++t) {
    v16bf kb = ldsFrag(&Ks[0][0], 34, t * 16, 0, lane);
    sacc[t] = __builtin_amdgcn_wmma_f32_16x16x32_bf16(
        false, qa, false, kb, (short)0, vzero, false, false);
  }
  // scale + rel-pos bias + shift mask -> Sf
#pragma unroll
  for (int t = 0; t < 4; ++t)
#pragma unroll
    for (int e = 0; e < 8; ++e) {
      int n = wave * 16 + e + hlf * 8;
      int m = t * 16 + cL;
      float val;
      if (n < 49 && m < 49) {
        val = sacc[t][e] * SCALE_ATTN;
        int iy = n / 7, ix = n % 7, jy = m / 7, jx = m % 7;
        int idx = (iy - jy + 6) * 13 + (ix - jx + 6);
        val += bias_table[idx * 8 + head];
        int rn = shift_region(wy * 7 + iy) * 3 + shift_region(wx * 7 + ix);
        int rm = shift_region(wy * 7 + jy) * 3 + shift_region(wx * 7 + jx);
        if (rn != rm) val -= 100.f;
      } else {
        val = -1e30f;
      }
      Sf[n][m] = val;
    }
  __syncthreads();

  // ---- softmax over 49 real cols, one thread per row ----
  if (tid < 64) {
    int n = tid;
    if (n < 49) {
      float mx = -1e30f;
      for (int m = 0; m < 49; ++m) mx = fmaxf(mx, Sf[n][m]);
      float sum = 0.f;
      for (int m = 0; m < 49; ++m) {
        float e = expf(Sf[n][m] - mx);
        Sf[n][m] = e;
        sum += e;
      }
      float inv = 1.f / sum;
      for (int m = 0; m < 49; ++m) Ps[n][m] = (bf16)(Sf[n][m] * inv);
      for (int m = 49; m < 64; ++m) Ps[n][m] = (bf16)0.f;
    } else {
      for (int m = 0; m < 64; ++m) Ps[n][m] = (bf16)0.f;
    }
  }
  __syncthreads();

  // ---- O = P * V : 2 K-steps of 32 tokens, 2 col tiles (HD=32) ----
  v8f oacc[2] = {vzero, vzero};
#pragma unroll
  for (int ks = 0; ks < 2; ++ks) {
    v16bf pa = ldsFrag(&Ps[0][0], 72, wave * 16, ks * 32, lane);
#pragma unroll
    for (int t = 0; t < 2; ++t) {
      v16bf vb = ldsFrag(&Vt[0][0], 72, t * 16, ks * 32, lane);
      oacc[t] = __builtin_amdgcn_wmma_f32_16x16x32_bf16(
          false, pa, false, vb, (short)0, oacc[t], false, false);
    }
  }
#pragma unroll
  for (int t = 0; t < 2; ++t)
#pragma unroll
    for (int e = 0; e < 8; ++e) {
      int n  = wave * 16 + e + hlf * 8;
      int hd = t * 16 + cL;
      if (n < 49)
        out[(size_t)(win * 49 + n) * 256 + head * 32 + hd] = (bf16)oacc[t][e];
    }
}

// -------------------------------------------------------------------------
extern "C" void kernel_launch(void* const* d_in, const int* in_sizes, int n_in,
                              void* d_out, int out_size, void* d_ws, size_t ws_size,
                              hipStream_t stream)
{
  (void)in_sizes; (void)n_in; (void)out_size; (void)ws_size;
  const float* x      = (const float*)d_in[0];
  const float* g1     = (const float*)d_in[1];
  const float* beta1  = (const float*)d_in[2];
  const float* w_qkv  = (const float*)d_in[3];
  const float* b_qkv  = (const float*)d_in[4];
  const float* btab   = (const float*)d_in[5];
  const float* w_proj = (const float*)d_in[6];
  const float* b_proj = (const float*)d_in[7];
  const float* g2     = (const float*)d_in[8];
  const float* beta2  = (const float*)d_in[9];
  const float* w_mlp1 = (const float*)d_in[10];
  const float* b_mlp1 = (const float*)d_in[11];
  const float* w_mlp2 = (const float*)d_in[12];
  const float* b_mlp2 = (const float*)d_in[13];
  float* out = (float*)d_out;

  char* ws = (char*)d_ws;
  bf16* xw     = (bf16*)ws;                                 // 200704*256 bf16
  bf16* qkv    = (bf16*)(ws + (size_t)TOKENS * 256 * 2);    // 200704*768 bf16 (reused: hidden 200704*1024)
  bf16* attnout = xw;   // xw consumed by QKV GEMM before attention writes
  bf16* yln    = xw;    // attnout consumed by proj before LN2 writes
  bf16* hidden = qkv;   // qkv consumed by attention before MLP1 writes

  // 1) LN1 + shift + window partition (fp32 -> bf16)
  ln_kernel<<<TOKENS / 8, 256, 0, stream>>>(x, g1, beta1, xw, 1);
  // 2) QKV GEMM [200704,256]x[256,768] -> q/k/v per (win,head)
  gemm_bf16_kernel<0><<<dim3(TOKENS / 128, 12), 256, 0, stream>>>(
      xw, w_qkv, b_qkv, qkv, nullptr, nullptr, TOKENS, 768, 256);
  // 3) Windowed attention: 4096 windows * 8 heads
  attn_kernel<<<4096 * 8, 128, 0, stream>>>(qkv, btab, attnout);
  // 4) Proj GEMM + window reverse + unshift + residual -> d_out (fp32)
  gemm_bf16_kernel<1><<<dim3(TOKENS / 128, 4), 256, 0, stream>>>(
      attnout, w_proj, b_proj, nullptr, out, x, TOKENS, 256, 256);
  // 5) LN2 (reads d_out) -> bf16
  ln_kernel<<<TOKENS / 8, 256, 0, stream>>>(out, g2, beta2, yln, 0);
  // 6) MLP1 GEMM + exact GELU -> bf16 hidden [200704,1024]
  gemm_bf16_kernel<2><<<dim3(TOKENS / 128, 16), 256, 0, stream>>>(
      yln, w_mlp1, b_mlp1, hidden, nullptr, nullptr, TOKENS, 1024, 256);
  // 7) MLP2 GEMM + residual accumulate into d_out
  gemm_bf16_kernel<3><<<dim3(TOKENS / 128, 4), 256, 0, stream>>>(
      hidden, w_mlp2, b_mlp2, nullptr, out, nullptr, TOKENS, 256, 1024);
}
